// Pointnet_Backbone_8323646620268
// MI455X (gfx1250) — compile-verified, matrix operand order checked by reading, not run on hardware
//
#include <hip/hip_runtime.h>

// ---------------------------------------------------------------------------
// PointNet++ backbone for MI455X (gfx1250, wave32, WMMA).
//  - FPS: 1 workgroup/batch, points + running min-dist cached in LDS.
//  - Ball query: thread per query, first-32-in-index-order (matches jnp.sort).
//  - Grouping: zero-padded f16 rows (Kp multiple of 32) for guard-free WMMA.
//  - MLP GEMMs: v_wmma_f32_16x16x32_f16, 16x64 strip per wave (4 n-tiles),
//    compile-time K => fully unrolled, software-pipelined (no register moves).
//  - Max-pool over 32 neighbors; final einsum('bsc,oc->bos') via WMMA to f32.
// ---------------------------------------------------------------------------

typedef __attribute__((ext_vector_type(16))) _Float16 v16h;
typedef __attribute__((ext_vector_type(8)))  float    v8f;
typedef _Float16 f16;

#define BATCH 8
#define NT 4  // n-tiles (16 cols each) per wave

// ---- WMMA fragment load: 16-bit A/B layout, 16x32 tile, row-major source ---
// lane<16  holds row (lane&15), K in {0..7} U {16..23}
// lane>=16 holds row (lane&15), K in {8..15} U {24..31}
__device__ __forceinline__ v16h load_frag(const f16* __restrict__ rowptr, int half) {
  union { v16h h; unsigned int u[8]; } f;
#pragma unroll
  for (int g = 0; g < 8; ++g) {
    const int k = (g < 4 ? 2 * g : 2 * g + 8) + 8 * half;
    f.u[g] = *(const unsigned int*)(rowptr + k);  // pair (k, k+1), dword aligned
  }
  return f.h;
}

__device__ __forceinline__ v8f wmma_step(v16h a, v16h b, v8f c) {
  return __builtin_amdgcn_wmma_f32_16x16x32_f16(
      /*neg_a=*/false, a, /*neg_b=*/false, b,
      /*c_mod=*/(short)0, c, /*reuse_a=*/false, /*reuse_b=*/false);
}

// Fully-unrolled, double-buffered K loop. KP is compile-time so the
// cur<-nxt rotation is renamed away (no v_mov chains) and loads for step
// k+32 are scheduled ahead of the WMMAs for step k.
template <int KP>
__device__ __forceinline__ void gemm_core(const f16* __restrict__ aRow,
                                          const f16* __restrict__ bRow,
                                          int half, v8f acc[NT]) {
  v16h aCur = load_frag(aRow, half);
  v16h bCur[NT];
#pragma unroll
  for (int t = 0; t < NT; ++t) bCur[t] = load_frag(bRow + (size_t)t * 16 * KP, half);

#pragma unroll
  for (int k0 = 32; k0 < KP; k0 += 32) {
    if (k0 + 128 <= KP) __builtin_prefetch(aRow + k0 + 96, 0, 3);  // global_prefetch_b8
    v16h aNxt = load_frag(aRow + k0, half);
    v16h bNxt[NT];
#pragma unroll
    for (int t = 0; t < NT; ++t) bNxt[t] = load_frag(bRow + (size_t)t * 16 * KP + k0, half);
#pragma unroll
    for (int t = 0; t < NT; ++t) acc[t] = wmma_step(aCur, bCur[t], acc[t]);
    aCur = aNxt;
#pragma unroll
    for (int t = 0; t < NT; ++t) bCur[t] = bNxt[t];
  }
#pragma unroll
  for (int t = 0; t < NT; ++t) acc[t] = wmma_step(aCur, bCur[t], acc[t]);
}

// ---------------------------- FPS --------------------------------------------
__global__ void fps_kernel(const float* __restrict__ xyz, int xyzStride,
                           float* __restrict__ newXyz, int N, int S) {
  extern __shared__ float smem[];
  float* sd = smem;          // N  running min distance
  float* sx = smem + N;      // N
  float* sy = smem + 2 * N;  // N
  float* sz = smem + 3 * N;  // N
  __shared__ float redV[8];
  __shared__ int   redI[8];
  __shared__ int   sFar;

  const int b   = blockIdx.x;
  const int tid = threadIdx.x;
  const float* base = xyz + (size_t)b * N * xyzStride;
  for (int i = tid; i < N; i += blockDim.x) {
    sd[i] = 1e10f;
    sx[i] = base[i * xyzStride + 0];
    sy[i] = base[i * xyzStride + 1];
    sz[i] = base[i * xyzStride + 2];
  }
  if (tid == 0) sFar = 0;
  __syncthreads();

  const int lane = tid & 31;
  const int wid  = tid >> 5;
  const int nw   = blockDim.x >> 5;

  for (int s = 0; s < S; ++s) {
    const int far = sFar;
    const float cx = sx[far], cy = sy[far], cz = sz[far];
    if (tid == 0) {
      float* o = newXyz + ((size_t)b * S + s) * 3;
      o[0] = cx; o[1] = cy; o[2] = cz;
    }
    float bv = -1.0f; int bi = 0;
    for (int i = tid; i < N; i += blockDim.x) {
      const float dx = sx[i] - cx, dy = sy[i] - cy, dz = sz[i] - cz;
      const float d  = dx * dx + dy * dy + dz * dz;
      const float nd = fminf(sd[i], d);
      sd[i] = nd;
      if (nd > bv) { bv = nd; bi = i; }
    }
    // wave32 argmax reduction (ties -> smaller index, like jnp.argmax)
    for (int off = 16; off > 0; off >>= 1) {
      const float ov = __shfl_down(bv, (unsigned)off, 32);
      const int   oi = __shfl_down(bi, (unsigned)off, 32);
      if (ov > bv || (ov == bv && oi < bi)) { bv = ov; bi = oi; }
    }
    if (lane == 0) { redV[wid] = bv; redI[wid] = bi; }
    __syncthreads();
    if (tid == 0) {
      float v = redV[0]; int idx = redI[0];
      for (int w = 1; w < nw; ++w)
        if (redV[w] > v || (redV[w] == v && redI[w] < idx)) { v = redV[w]; idx = redI[w]; }
      sFar = idx;
    }
    __syncthreads();
  }
}

// ---------------------------- Ball query -------------------------------------
__global__ void ballquery_kernel(const float* __restrict__ xyz, int xyzStride,
                                 const float* __restrict__ newXyz,
                                 int* __restrict__ nidx,
                                 int N, int S, float r2) {
  const int t = blockIdx.x * blockDim.x + threadIdx.x;
  if (t >= BATCH * S) return;
  const int b = t / S, s = t % S;
  const float* base = xyz + (size_t)b * N * xyzStride;
  const float* c = newXyz + ((size_t)b * S + s) * 3;
  const float cx = c[0], cy = c[1], cz = c[2];
  int* out = nidx + (size_t)t * 32;
  int cnt = 0, first = 0;
  for (int i = 0; i < N && cnt < 32; ++i) {
    const float dx = base[i * xyzStride + 0] - cx;
    const float dy = base[i * xyzStride + 1] - cy;
    const float dz = base[i * xyzStride + 2] - cz;
    if (dx * dx + dy * dy + dz * dz < r2) {
      if (cnt == 0) first = i;
      out[cnt++] = i;
    }
  }
  for (; cnt < 32; ++cnt) out[cnt] = first;  // centroid always qualifies -> cnt>=1
}

// ---------------------------- Grouping (zero-padded f16 rows) ---------------
__global__ void group_kernel(const float* __restrict__ xyz, int xyzStride,
                             const float* __restrict__ newXyz,
                             const float* __restrict__ featF32, int featF32Stride,
                             const f16* __restrict__ featF16,
                             int Cf, const int* __restrict__ nidx,
                             f16* __restrict__ h0, int N, int S, int Kp) {
  const int row = blockIdx.x;           // (b*S + s)*32 + j
  const int bs = row >> 5;
  const int b = bs / S, s = bs % S;
  const int idx = nidx[row];
  f16* dst = h0 + (size_t)row * Kp;
  for (int c = threadIdx.x; c < Kp; c += blockDim.x) {
    float v;
    if (c < 3) {
      v = xyz[((size_t)b * N + idx) * xyzStride + c] - newXyz[((size_t)b * S + s) * 3 + c];
    } else if (c < 3 + Cf) {
      const int cc = c - 3;
      v = featF32 ? featF32[((size_t)b * N + idx) * featF32Stride + cc]
                  : (float)featF16[((size_t)b * N + idx) * Cf + cc];
    } else {
      v = 0.0f;  // K padding
    }
    dst[c] = (f16)v;
  }
}

// ---------------------------- Weight fp32 -> padded f16 ----------------------
__global__ void convw_kernel(const float* __restrict__ W, f16* __restrict__ W16,
                             int Cout, int Cin, int Kp) {
  const int t = blockIdx.x * blockDim.x + threadIdx.x;
  if (t >= Cout * Kp) return;
  const int n = t / Kp, k = t % Kp;
  W16[t] = (f16)(k < Cin ? W[n * Cin + k] : 0.0f);
}

// ---------------------------- WMMA GEMM + bias + ReLU ------------------------
// D[R x N] = relu(A[R x KP] * Bw[N x KP]^T + bias).
// Each wave: one 16x64 output strip (NT=4 n-tiles), fully-unrolled K.
template <int KP>
__global__ void __launch_bounds__(256)
gemm_relu_f16(const f16* __restrict__ A, const f16* __restrict__ Bw,
              const float* __restrict__ bias, f16* __restrict__ D,
              int R, int N) {
  const int lane  = threadIdx.x & 31;
  const int wave  = threadIdx.x >> 5;
  const int mTile = blockIdx.x * 8 + wave;
  const int nBase = blockIdx.y * (16 * NT);
  const int half  = lane >> 4;
  const int l16   = lane & 15;

  const f16* aRow = A  + (size_t)(mTile * 16 + l16) * KP;
  const f16* bRow = Bw + (size_t)(nBase + l16) * KP;

  v8f acc[NT];
#pragma unroll
  for (int t = 0; t < NT; ++t) {
    const float bv = bias[nBase + t * 16 + l16];
    acc[t] = (v8f){ bv, bv, bv, bv, bv, bv, bv, bv };
  }

  gemm_core<KP>(aRow, bRow, half, acc);

  // store with fused ReLU (f16, row-major)
#pragma unroll
  for (int t = 0; t < NT; ++t) {
    f16* d0 = D + (size_t)(mTile * 16) * N + nBase + t * 16 + l16;
#pragma unroll
    for (int r = 0; r < 8; ++r) {
      const int m = r + 8 * half;
      const float v = acc[t][r];
      d0[(size_t)m * N] = (f16)(v > 0.0f ? v : 0.0f);
    }
  }
}

// ---------------------------- Final WMMA GEMM: einsum('bsc,oc->bos') + bf ----
__global__ void __launch_bounds__(256)
gemm_final(const f16* __restrict__ A, const f16* __restrict__ Bw,
           const float* __restrict__ bias, float* __restrict__ out, int Sdim) {
  constexpr int KP = 256;
  const int lane  = threadIdx.x & 31;
  const int wave  = threadIdx.x >> 5;
  const int mTile = blockIdx.x * 8 + wave;
  const int nBase = blockIdx.y * (16 * NT);
  const int half  = lane >> 4;
  const int l16   = lane & 15;

  const f16* aRow = A  + (size_t)(mTile * 16 + l16) * KP;
  const f16* bRow = Bw + (size_t)(nBase + l16) * KP;

  v8f acc[NT];
#pragma unroll
  for (int t = 0; t < NT; ++t) {
    const float bv = bias[nBase + t * 16 + l16];
    acc[t] = (v8f){ bv, bv, bv, bv, bv, bv, bv, bv };
  }

  gemm_core<KP>(aRow, bRow, half, acc);

#pragma unroll
  for (int t = 0; t < NT; ++t) {
    const int o = nBase + t * 16 + l16;
#pragma unroll
    for (int r = 0; r < 8; ++r) {
      const int m   = r + 8 * half;
      const int row = mTile * 16 + m;          // row = b*Sdim + s
      const int b   = row / Sdim;
      const int s   = row % Sdim;
      out[((size_t)b * 256 + o) * Sdim + s] = acc[t][r];
    }
  }
}

// ---------------------------- Max-pool over 32 neighbors ---------------------
__global__ void maxpool_kernel(const f16* __restrict__ H, f16* __restrict__ feats,
                               int C, int total) {
  const int t = blockIdx.x * blockDim.x + threadIdx.x;
  if (t >= total) return;
  const int bs = t / C, c = t % C;
  const f16* p = H + ((size_t)bs * 32) * C + c;
  float m = (float)p[0];
  for (int j = 1; j < 32; ++j) m = fmaxf(m, (float)p[(size_t)j * C]);
  feats[t] = (f16)m;
}

// ---------------------------------------------------------------------------
extern "C" void kernel_launch(void* const* d_in, const int* in_sizes, int n_in,
                              void* d_out, int out_size, void* d_ws, size_t ws_size,
                              hipStream_t stream) {
  (void)in_sizes; (void)n_in; (void)out_size;

  const float* pc  = (const float*)d_in[0];
  // d_in[1..3] = n1,n2,n3 scalars (fixed 1024/512/256 per reference setup)
  const float* w1a = (const float*)d_in[4];  const float* b1a = (const float*)d_in[5];
  const float* w1b = (const float*)d_in[6];  const float* b1b = (const float*)d_in[7];
  const float* w1c = (const float*)d_in[8];  const float* b1c = (const float*)d_in[9];
  const float* w2a = (const float*)d_in[10]; const float* b2a = (const float*)d_in[11];
  const float* w2b = (const float*)d_in[12]; const float* b2b = (const float*)d_in[13];
  const float* w2c = (const float*)d_in[14]; const float* b2c = (const float*)d_in[15];
  const float* w3a = (const float*)d_in[16]; const float* b3a = (const float*)d_in[17];
  const float* w3b = (const float*)d_in[18]; const float* b3b = (const float*)d_in[19];
  const float* w3c = (const float*)d_in[20]; const float* b3c = (const float*)d_in[21];
  const float* wf  = (const float*)d_in[22]; const float* bf  = (const float*)d_in[23];

  // ---- workspace arena (~116 MB), same offsets every call -------------------
  char* ws = (char*)d_ws;
  size_t off = 0;
  auto bump = [&](size_t bytes) {
    size_t o = off; off += (bytes + 255) & ~(size_t)255; return o;
  };
  float* xyz1   = (float*)(ws + bump((size_t)BATCH * 1024 * 3 * 4));
  float* xyz2   = (float*)(ws + bump((size_t)BATCH * 512 * 3 * 4));
  f16*   feats1 = (f16*)  (ws + bump((size_t)BATCH * 1024 * 128 * 2));
  f16*   feats2 = (f16*)  (ws + bump((size_t)BATCH * 512 * 256 * 2));
  f16*   feats3 = (f16*)  (ws + bump((size_t)BATCH * 256 * 256 * 2));
  int*   nidx   = (int*)  (ws + bump((size_t)BATCH * 1024 * 32 * 4));
  f16* w16_1a = (f16*)(ws + bump((size_t)64  * 32  * 2));
  f16* w16_1b = (f16*)(ws + bump((size_t)64  * 64  * 2));
  f16* w16_1c = (f16*)(ws + bump((size_t)128 * 64  * 2));
  f16* w16_2a = (f16*)(ws + bump((size_t)128 * 160 * 2));
  f16* w16_2b = (f16*)(ws + bump((size_t)128 * 128 * 2));
  f16* w16_2c = (f16*)(ws + bump((size_t)256 * 128 * 2));
  f16* w16_3a = (f16*)(ws + bump((size_t)256 * 288 * 2));
  f16* w16_3b = (f16*)(ws + bump((size_t)256 * 256 * 2));
  f16* w16_3c = (f16*)(ws + bump((size_t)256 * 256 * 2));
  f16* w16_f  = (f16*)(ws + bump((size_t)256 * 256 * 2));
  f16* bufA = (f16*)(ws + bump((size_t)131072 * 160 * 2));  // 42 MB ping
  f16* bufB = (f16*)(ws + bump((size_t)262144 * 128 * 2));  // 67 MB pong
  (void)ws_size;

  float* outXyz  = (float*)d_out;           // (8,256,3)
  float* outFeat = (float*)d_out + BATCH * 256 * 3;

  // ---- weights -> padded f16 ------------------------------------------------
  struct { const float* w; f16* w16; int co, ci, kp; } wc[10] = {
    {w1a, w16_1a,  64,   6,  32}, {w1b, w16_1b,  64,  64,  64},
    {w1c, w16_1c, 128,  64,  64}, {w2a, w16_2a, 128, 131, 160},
    {w2b, w16_2b, 128, 128, 128}, {w2c, w16_2c, 256, 128, 128},
    {w3a, w16_3a, 256, 259, 288}, {w3b, w16_3b, 256, 256, 256},
    {w3c, w16_3c, 256, 256, 256}, {wf,  w16_f,  256, 256, 256}};
  for (int i = 0; i < 10; ++i) {
    const int tot = wc[i].co * wc[i].kp;
    convw_kernel<<<(tot + 255) / 256, 256, 0, stream>>>(wc[i].w, wc[i].w16,
                                                        wc[i].co, wc[i].ci, wc[i].kp);
  }

  // templated GEMM dispatch on compile-time K
  auto gemm = [&](const f16* A, const f16* Bw, const float* bias, f16* D,
                  int R, int Kp, int N) {
    dim3 grid(R / 128, N / 64);
    switch (Kp) {
      case 32:  gemm_relu_f16<32> <<<grid, 256, 0, stream>>>(A, Bw, bias, D, R, N); break;
      case 64:  gemm_relu_f16<64> <<<grid, 256, 0, stream>>>(A, Bw, bias, D, R, N); break;
      case 128: gemm_relu_f16<128><<<grid, 256, 0, stream>>>(A, Bw, bias, D, R, N); break;
      case 160: gemm_relu_f16<160><<<grid, 256, 0, stream>>>(A, Bw, bias, D, R, N); break;
      case 256: gemm_relu_f16<256><<<grid, 256, 0, stream>>>(A, Bw, bias, D, R, N); break;
      case 288: gemm_relu_f16<288><<<grid, 256, 0, stream>>>(A, Bw, bias, D, R, N); break;
    }
  };

  // ---- Stage 1: N=4096 (stride 6) -> S=1024, r=0.3, 6->64->64->128 ----------
  {
    const int N = 4096, S = 1024, R = BATCH * S * 32;
    fps_kernel<<<BATCH, 256, 4 * N * sizeof(float), stream>>>(pc, 6, xyz1, N, S);
    ballquery_kernel<<<(BATCH * S + 255) / 256, 256, 0, stream>>>(pc, 6, xyz1, nidx, N, S, 0.09f);
    group_kernel<<<BATCH * S * 32, 32, 0, stream>>>(pc, 6, xyz1, pc + 3, 6, nullptr,
                                                    3, nidx, bufA, N, S, 32);
    gemm(bufA, w16_1a, b1a, bufB, R, 32, 64);
    gemm(bufB, w16_1b, b1b, bufA, R, 64, 64);
    gemm(bufA, w16_1c, b1c, bufB, R, 64, 128);
    maxpool_kernel<<<(BATCH * S * 128 + 255) / 256, 256, 0, stream>>>(bufB, feats1, 128,
                                                                      BATCH * S * 128);
  }

  // ---- Stage 2: N=1024 -> S=512, r=0.5, 131->128->128->256 ------------------
  {
    const int N = 1024, S = 512, R = BATCH * S * 32;
    fps_kernel<<<BATCH, 256, 4 * N * sizeof(float), stream>>>(xyz1, 3, xyz2, N, S);
    ballquery_kernel<<<(BATCH * S + 255) / 256, 256, 0, stream>>>(xyz1, 3, xyz2, nidx, N, S, 0.25f);
    group_kernel<<<BATCH * S * 32, 160, 0, stream>>>(xyz1, 3, xyz2, nullptr, 0, feats1,
                                                     128, nidx, bufA, N, S, 160);
    gemm(bufA, w16_2a, b2a, bufB, R, 160, 128);
    gemm(bufB, w16_2b, b2b, bufA, R, 128, 128);
    gemm(bufA, w16_2c, b2c, bufB, R, 128, 256);
    maxpool_kernel<<<(BATCH * S * 256 + 255) / 256, 256, 0, stream>>>(bufB, feats2, 256,
                                                                      BATCH * S * 256);
  }

  // ---- Stage 3: N=512 -> S=256, r=0.7, 259->256->256->256; xyz -> d_out -----
  {
    const int N = 512, S = 256, R = BATCH * S * 32;
    fps_kernel<<<BATCH, 256, 4 * N * sizeof(float), stream>>>(xyz2, 3, outXyz, N, S);
    ballquery_kernel<<<(BATCH * S + 255) / 256, 256, 0, stream>>>(xyz2, 3, outXyz, nidx, N, S, 0.49f);
    group_kernel<<<BATCH * S * 32, 288, 0, stream>>>(xyz2, 3, outXyz, nullptr, 0, feats2,
                                                     256, nidx, bufA, N, S, 288);
    gemm(bufA, w16_3a, b3a, bufB, R, 288, 256);
    gemm(bufB, w16_3b, b3b, bufA, R, 256, 256);
    gemm(bufA, w16_3c, b3c, bufB, R, 256, 256);
    maxpool_kernel<<<(BATCH * S * 256 + 255) / 256, 256, 0, stream>>>(bufB, feats3, 256,
                                                                      BATCH * S * 256);
  }

  // ---- Final: feature[b,o,s] = wf @ feats3 + bf -----------------------------
  gemm_final<<<dim3((BATCH * 256) / 128, 4), 256, 0, stream>>>(feats3, w16_f, bf,
                                                               outFeat, 256);
}
